// MemoryEfficientAttnBlock3D_69827578298391
// MI455X (gfx1250) — matrix operand order repaired and checked by reading, not compiled, not run on hardware
//
#include <hip/hip_runtime.h>
#include <stdint.h>

// ---------------------------------------------------------------------------
// MemoryEfficientAttnBlock3D for MI455X (gfx1250, wave32, WMMA)
//   b=2, c=64, n=8192 tokens, 32 groups GroupNorm
// Flash-attention streaming: S (8192x8192) never materialized.
// bf16 WMMA (v_wmma_f32_16x16x32_bf16) for QKV proj, QK^T, PV, out proj.
// K/V tiles double-buffered in LDS via global_load_async_to_lds_b128.
// ---------------------------------------------------------------------------

typedef __attribute__((ext_vector_type(16))) __bf16 v16bf;
typedef __attribute__((ext_vector_type(8)))  float  v8f;

union FragB {
  uint4 u[2];
  v16bf v;
};
static_assert(sizeof(FragB) == 32, "frag size");

#define WMMA_BF16(A, B, C) \
  __builtin_amdgcn_wmma_f32_16x16x32_bf16(false, (A), false, (B), (short)0, (C), false, false)

#define N_TOK 8192
#define C_DIM 64

// --- CDNA5 async copy global -> LDS (16B per lane), tracked by ASYNCcnt ----
__device__ __forceinline__ void async_copy_b128(uint32_t lds_off, uint64_t gaddr) {
  asm volatile("global_load_async_to_lds_b128 %0, %1, off"
               :: "v"(lds_off), "v"(gaddr)
               : "memory");
}
__device__ __forceinline__ void wait_async_all() {
  asm volatile("s_wait_asynccnt 0x0" ::: "memory");
}

// ---------------------------------------------------------------------------
// Kernel 1: GroupNorm statistics. One block per (batch, group).
// Group g covers channels {2g, 2g+1} -> 16384 contiguous floats.
// ---------------------------------------------------------------------------
__global__ void k_stats(const float* __restrict__ x, float* __restrict__ stats) {
  __shared__ float ss[256];
  __shared__ float ss2[256];
  const int tid = threadIdx.x;
  const int bg = blockIdx.x;           // 0..63
  const int batch = bg >> 5;
  const int g = bg & 31;
  const float* p = x + ((size_t)(batch * 64 + g * 2)) * N_TOK;
  float s = 0.f, s2 = 0.f;
  for (int idx = tid; idx < 2 * N_TOK; idx += 256) {
    float v = p[idx];
    s += v;
    s2 += v * v;
  }
  ss[tid] = s;
  ss2[tid] = s2;
  __syncthreads();
  for (int off = 128; off > 0; off >>= 1) {
    if (tid < off) {
      ss[tid] += ss[tid + off];
      ss2[tid] += ss2[tid + off];
    }
    __syncthreads();
  }
  if (tid == 0) {
    const float inv_n = 1.0f / 16384.0f;
    float mean = ss[0] * inv_n;
    float var = ss2[0] * inv_n - mean * mean;
    stats[bg * 2 + 0] = mean;
    stats[bg * 2 + 1] = rsqrtf(var + 1e-6f);
  }
}

// ---------------------------------------------------------------------------
// Kernel 1b: convert the four 64x64 weight matrices to bf16 (row-major o,c).
// ---------------------------------------------------------------------------
__global__ void k_wcvt(const float* __restrict__ wq, const float* __restrict__ wk,
                       const float* __restrict__ wv, const float* __restrict__ wp,
                       __bf16* __restrict__ wbf) {
  int idx = blockIdx.x * 256 + threadIdx.x;   // 0..16383
  int w = idx >> 12, j = idx & 4095;
  const float* src = (w == 0) ? wq : (w == 1) ? wk : (w == 2) ? wv : wp;
  wbf[idx] = (__bf16)src[j];
}

// ---------------------------------------------------------------------------
// Kernel 2: GroupNorm apply + QKV projection via WMMA.
// Block = 256 threads (8 waves), 64 tokens per block.
// Phase 1: load x (c-major), normalize, store bf16 token-major tile to LDS.
// Phase 2: per wave, D[token, o] = hn^T * W^T via 2 chained WMMAs (K=64).
// Q/K written (b, n, c) bf16 (Q pre-scaled by 1/8); V written (b, c, n) bf16.
// ---------------------------------------------------------------------------
__global__ void k_norm_qkv(const float* __restrict__ x, const float* __restrict__ gamma,
                           const float* __restrict__ beta, const float* __restrict__ bq,
                           const float* __restrict__ bk, const float* __restrict__ bv,
                           const float* __restrict__ stats, const __bf16* __restrict__ wbf,
                           __bf16* __restrict__ qbf, __bf16* __restrict__ kbf,
                           __bf16* __restrict__ vbf) {
  __shared__ __align__(16) __bf16 hn[64 * 64];   // [token][c], 8 KB
  const int tid = threadIdx.x;
  const int batch = blockIdx.x >> 7;
  const int t0 = (blockIdx.x & 127) * 64;

  // ---- phase 1: normalize + transpose into LDS ----
  {
    const int c = tid >> 2;
    const int jg = tid & 3;
    const int g = c >> 1;
    const float mean = stats[(batch * 32 + g) * 2 + 0];
    const float inv  = stats[(batch * 32 + g) * 2 + 1];
    const float sc = inv * gamma[c];
    const float sh = beta[c] - mean * sc;
    const float4* xp = (const float4*)(x + ((size_t)(batch * 64 + c)) * N_TOK + t0 + jg * 16);
#pragma unroll
    for (int k4 = 0; k4 < 4; ++k4) {
      float4 v = xp[k4];
      int tk = jg * 16 + k4 * 4;
      hn[(tk + 0) * 64 + c] = (__bf16)(v.x * sc + sh);
      hn[(tk + 1) * 64 + c] = (__bf16)(v.y * sc + sh);
      hn[(tk + 2) * 64 + c] = (__bf16)(v.z * sc + sh);
      hn[(tk + 3) * 64 + c] = (__bf16)(v.w * sc + sh);
    }
  }
  __syncthreads();

  // ---- phase 2: WMMA projections ----
  const int lane = tid & 31;
  const int wave = tid >> 5;
  const int i = lane & 15;
  const int hi = lane >> 4;
  const int tt = wave & 3;       // token sub-tile
  const int oh = wave >> 2;      // output-channel half

  // A fragments: rows = tokens (LDS, 128 B/row), K = c, split K=64 -> 2 frags
  const char* hrow = (const char*)hn + (size_t)(tt * 16 + i) * 128;
  FragB A0, A1;
  A0.u[0] = *(const uint4*)(hrow + 32 * hi);
  A0.u[1] = *(const uint4*)(hrow + 32 * hi + 16);
  A1.u[0] = *(const uint4*)(hrow + 64 + 32 * hi);
  A1.u[1] = *(const uint4*)(hrow + 64 + 32 * hi + 16);

#pragma unroll
  for (int out = 0; out < 3; ++out) {
    const __bf16* W = wbf + out * 4096;
    const float* bias = (out == 0) ? bq : (out == 1) ? bk : bv;
#pragma unroll
    for (int oti = 0; oti < 2; ++oti) {
      const int ot = oh * 2 + oti;
      const int o = ot * 16 + i;
      const char* wrow = (const char*)W + (size_t)o * 128;   // row o of W (c contiguous)
      FragB B0, B1;
      B0.u[0] = *(const uint4*)(wrow + 32 * hi);
      B0.u[1] = *(const uint4*)(wrow + 32 * hi + 16);
      B1.u[0] = *(const uint4*)(wrow + 64 + 32 * hi);
      B1.u[1] = *(const uint4*)(wrow + 64 + 32 * hi + 16);
      v8f acc = {0.f, 0.f, 0.f, 0.f, 0.f, 0.f, 0.f, 0.f};
      acc = WMMA_BF16(A0.v, B0.v, acc);
      acc = WMMA_BF16(A1.v, B1.v, acc);
      const float bs = bias[o];
#pragma unroll
      for (int r = 0; r < 8; ++r) {
        const int token = t0 + tt * 16 + r + 8 * hi;
        const float val = acc[r] + bs;
        if (out == 0)
          qbf[((size_t)(batch * N_TOK + token)) * 64 + o] = (__bf16)(val * 0.125f);
        else if (out == 1)
          kbf[((size_t)(batch * N_TOK + token)) * 64 + o] = (__bf16)val;
        else
          vbf[((size_t)(batch * 64 + o)) * N_TOK + token] = (__bf16)val;
      }
    }
  }
}

// ---------------------------------------------------------------------------
// Kernel 3: flash attention. Block = 128 threads (4 waves), 64 queries/block.
// Each wave owns 16 queries; loops over 256 KV tiles of 32 keys.
// S^T = K·Q^T so the exp'd P tile feeds the PV WMMA as the B operand with
// only shfl_xor(16) lane exchanges. K/V tiles double-buffered in LDS via
// async-to-LDS copies (ASYNCcnt).
// ---------------------------------------------------------------------------
__global__ void k_attn(const __bf16* __restrict__ qbf, const __bf16* __restrict__ kbf,
                       const __bf16* __restrict__ vbf, float* __restrict__ obuf) {
  __shared__ __align__(16) char sm[16384];   // K[2][4KB] then V[2][4KB]
  const int tid = threadIdx.x;               // 0..127
  const int batch = blockIdx.x >> 7;
  const int qblk = (blockIdx.x & 127) * 64;
  const int lane = tid & 31;
  const int wave = tid >> 5;
  const int i = lane & 15;
  const int hi = lane >> 4;
  const int q0 = qblk + wave * 16;

  // Q as B-operand fragments (held in registers for the whole kernel)
  const char* qrow = (const char*)qbf + ((size_t)(batch * N_TOK + q0 + i)) * 128;
  FragB bQ0, bQ1;
  bQ0.u[0] = *(const uint4*)(qrow + 32 * hi);
  bQ0.u[1] = *(const uint4*)(qrow + 32 * hi + 16);
  bQ1.u[0] = *(const uint4*)(qrow + 64 + 32 * hi);
  bQ1.u[1] = *(const uint4*)(qrow + 64 + 32 * hi + 16);

  // async-copy roles (32B of K + 32B of V tile data per thread, each a pair of b128s)
  const char* kg = (const char*)kbf + (size_t)batch * N_TOK * 128;        // (n,c) bf16
  const char* vg = (const char*)vbf + (size_t)batch * 64 * N_TOK * 2;     // (c,n) bf16
  const uint32_t k_t_off = (uint32_t)(tid * 32);          // within 4 KB K tile
  const int vc = tid >> 1, vh = tid & 1;
  const size_t v_g_off = ((size_t)vc * N_TOK + vh * 16) * 2;
  const uint32_t v_t_off = (uint32_t)(vc * 64 + vh * 32);
  const uint32_t lds_base = (uint32_t)(uintptr_t)sm;

  auto issue_tile = [&](int buf, int it) {
    uint64_t kga = (uint64_t)(uintptr_t)(kg + (size_t)it * 4096 + k_t_off);
    uint32_t klds = lds_base + (uint32_t)(buf * 4096) + k_t_off;
    async_copy_b128(klds, kga);
    async_copy_b128(klds + 16, kga + 16);
    uint64_t vga = (uint64_t)(uintptr_t)(vg + (size_t)it * 64 + v_g_off);
    uint32_t vlds = lds_base + 8192u + (uint32_t)(buf * 4096) + v_t_off;
    async_copy_b128(vlds, vga);
    async_copy_b128(vlds + 16, vga + 16);
  };

  float m = -1e30f, lsum = 0.f;
  v8f acc[4];
#pragma unroll
  for (int ct = 0; ct < 4; ++ct)
#pragma unroll
    for (int r = 0; r < 8; ++r) acc[ct][r] = 0.f;

  const int NIT = N_TOK / 32;
  issue_tile(0, 0);
  wait_async_all();
  __syncthreads();

  for (int it = 0; it < NIT; ++it) {
    const int cur = it & 1;
    if (it + 1 < NIT) issue_tile(cur ^ 1, it + 1);   // prefetch next tile
    const char* kt = sm + cur * 4096;
    const char* vt = sm + 8192 + cur * 4096;

    // ---- S^T tiles: D[j, i] = sum_c K[j,c] * Qs[i,c] (2 tiles x K=64) ----
    v8f s[2];
#pragma unroll
    for (int jt = 0; jt < 2; ++jt) {
      const char* krow = kt + (size_t)(jt * 16 + i) * 128;
      FragB aK0, aK1;
      aK0.u[0] = *(const uint4*)(krow + 32 * hi);
      aK0.u[1] = *(const uint4*)(krow + 32 * hi + 16);
      aK1.u[0] = *(const uint4*)(krow + 64 + 32 * hi);
      aK1.u[1] = *(const uint4*)(krow + 64 + 32 * hi + 16);
      v8f z = {0.f, 0.f, 0.f, 0.f, 0.f, 0.f, 0.f, 0.f};
      z = WMMA_BF16(aK0.v, bQ0.v, z);
      z = WMMA_BF16(aK1.v, bQ1.v, z);
      s[jt] = z;
    }

    // ---- online softmax along j (lane holds 16 of 32 j's for its query i) --
    float tmax = s[0][0];
#pragma unroll
    for (int r = 1; r < 8; ++r) tmax = fmaxf(tmax, s[0][r]);
#pragma unroll
    for (int r = 0; r < 8; ++r) tmax = fmaxf(tmax, s[1][r]);
    tmax = fmaxf(tmax, __shfl_xor(tmax, 16, 32));
    const float mnew = fmaxf(m, tmax);
    const float alpha = __expf(m - mnew);
    float p0[8], p1[8], rsum = 0.f;
#pragma unroll
    for (int r = 0; r < 8; ++r) { p0[r] = __expf(s[0][r] - mnew); rsum += p0[r]; }
#pragma unroll
    for (int r = 0; r < 8; ++r) { p1[r] = __expf(s[1][r] - mnew); rsum += p1[r]; }
    rsum += __shfl_xor(rsum, 16, 32);
    lsum = lsum * alpha + rsum;
    m = mnew;

    // ---- build P as B-operand fragment (cross-half exchange + bf16 pack) --
    float e0[8], e1[8];
#pragma unroll
    for (int r = 0; r < 8; ++r) e0[r] = __shfl_xor(p0[r], 16, 32);
#pragma unroll
    for (int r = 0; r < 8; ++r) e1[r] = __shfl_xor(p1[r], 16, 32);
    FragB bP;
#pragma unroll
    for (int e = 0; e < 16; ++e) {
      float sv = (hi == 0) ? ((e < 8) ? p0[e] : e0[e - 8])
                           : ((e < 8) ? e1[e] : p1[e - 8]);
      bP.v[e] = (__bf16)sv;
    }

    // ---- O[c, i] += V^T · P (4 channel tiles, K=32) ----
#pragma unroll
    for (int ct = 0; ct < 4; ++ct) {
      const char* vrow = vt + (size_t)(ct * 16 + i) * 64;
      FragB aV;
      aV.u[0] = *(const uint4*)(vrow + 32 * hi);
      aV.u[1] = *(const uint4*)(vrow + 32 * hi + 16);
#pragma unroll
      for (int r = 0; r < 8; ++r) acc[ct][r] *= alpha;
      acc[ct] = WMMA_BF16(aV.v, bP.v, acc[ct]);
    }

    wait_async_all();
    __syncthreads();
  }

  // ---- normalize + write O (b, c, n) fp32 ----
  const float invl = 1.0f / lsum;
#pragma unroll
  for (int ct = 0; ct < 4; ++ct) {
#pragma unroll
    for (int r = 0; r < 8; ++r) {
      const int c = ct * 16 + r + 8 * hi;
      obuf[((size_t)(batch * 64 + c)) * N_TOK + q0 + i] = acc[ct][r] * invl;
    }
  }
}

// ---------------------------------------------------------------------------
// Kernel 4: output projection (Wp) + bias + residual. Same scheme as kernel 2.
// ---------------------------------------------------------------------------
__global__ void k_proj_out(const float* __restrict__ x, const float* __restrict__ obuf,
                           const __bf16* __restrict__ wpbf, const float* __restrict__ bp,
                           float* __restrict__ out) {
  __shared__ __align__(16) __bf16 ot[64 * 64];
  const int tid = threadIdx.x;
  const int batch = blockIdx.x >> 7;
  const int t0 = (blockIdx.x & 127) * 64;

  {
    const int c = tid >> 2;
    const int jg = tid & 3;
    const float4* op = (const float4*)(obuf + ((size_t)(batch * 64 + c)) * N_TOK + t0 + jg * 16);
#pragma unroll
    for (int k4 = 0; k4 < 4; ++k4) {
      float4 v = op[k4];
      int tk = jg * 16 + k4 * 4;
      ot[(tk + 0) * 64 + c] = (__bf16)v.x;
      ot[(tk + 1) * 64 + c] = (__bf16)v.y;
      ot[(tk + 2) * 64 + c] = (__bf16)v.z;
      ot[(tk + 3) * 64 + c] = (__bf16)v.w;
    }
  }
  __syncthreads();

  const int lane = tid & 31;
  const int wave = tid >> 5;
  const int i = lane & 15;
  const int hi = lane >> 4;
  const int tt = wave & 3;
  const int oh = wave >> 2;

  const char* orow = (const char*)ot + (size_t)(tt * 16 + i) * 128;
  FragB A0, A1;
  A0.u[0] = *(const uint4*)(orow + 32 * hi);
  A0.u[1] = *(const uint4*)(orow + 32 * hi + 16);
  A1.u[0] = *(const uint4*)(orow + 64 + 32 * hi);
  A1.u[1] = *(const uint4*)(orow + 64 + 32 * hi + 16);

#pragma unroll
  for (int oti = 0; oti < 2; ++oti) {
    const int o = (oh * 2 + oti) * 16 + i;
    const char* wrow = (const char*)wpbf + (size_t)o * 128;
    FragB B0, B1;
    B0.u[0] = *(const uint4*)(wrow + 32 * hi);
    B0.u[1] = *(const uint4*)(wrow + 32 * hi + 16);
    B1.u[0] = *(const uint4*)(wrow + 64 + 32 * hi);
    B1.u[1] = *(const uint4*)(wrow + 64 + 32 * hi + 16);
    v8f acc = {0.f, 0.f, 0.f, 0.f, 0.f, 0.f, 0.f, 0.f};
    acc = WMMA_BF16(A0.v, B0.v, acc);
    acc = WMMA_BF16(A1.v, B1.v, acc);
    const float bs = bp[o];
#pragma unroll
    for (int r = 0; r < 8; ++r) {
      const int token = t0 + tt * 16 + r + 8 * hi;
      const size_t idx = ((size_t)(batch * 64 + o)) * N_TOK + token;
      out[idx] = x[idx] + acc[r] + bs;
    }
  }
}

// ---------------------------------------------------------------------------
// Host launcher. Workspace layout (bytes):
//   [0, 512)            group stats (mean, rsqrt(var+eps)) x 64
//   [1024, 33792)       weights bf16: wq|wk|wv|wp, 4 x 4096 elems
//   [64K, 64K+2M)       Q bf16 (b, n, c), pre-scaled by 1/sqrt(c)
//   [64K+2M, 64K+4M)    K bf16 (b, n, c)
//   [64K+4M, 64K+6M)    V bf16 (b, c, n)
//   [64K+6M, 64K+10M)   attention output O fp32 (b, c, n)
// ---------------------------------------------------------------------------
extern "C" void kernel_launch(void* const* d_in, const int* in_sizes, int n_in,
                              void* d_out, int out_size, void* d_ws, size_t ws_size,
                              hipStream_t stream) {
  const float* x     = (const float*)d_in[0];
  const float* gamma = (const float*)d_in[1];
  const float* beta  = (const float*)d_in[2];
  const float* wq    = (const float*)d_in[3];
  const float* bq    = (const float*)d_in[4];
  const float* wk    = (const float*)d_in[5];
  const float* bk    = (const float*)d_in[6];
  const float* wv    = (const float*)d_in[7];
  const float* bv    = (const float*)d_in[8];
  const float* wp    = (const float*)d_in[9];
  const float* bp    = (const float*)d_in[10];

  char* ws = (char*)d_ws;
  float*  stats = (float*)ws;
  __bf16* wbf   = (__bf16*)(ws + 1024);
  __bf16* qbf   = (__bf16*)(ws + (64 << 10));
  __bf16* kbf   = (__bf16*)(ws + (64 << 10) + (2 << 20));
  __bf16* vbf   = (__bf16*)(ws + (64 << 10) + (4 << 20));
  float*  obuf  = (float*) (ws + (64 << 10) + (6 << 20));

  k_stats<<<64, 256, 0, stream>>>(x, stats);
  k_wcvt<<<64, 256, 0, stream>>>(wq, wk, wv, wp, wbf);
  k_norm_qkv<<<256, 256, 0, stream>>>(x, gamma, beta, bq, bk, bv, stats, wbf, qbf, kbf, vbf);
  k_attn<<<256, 128, 0, stream>>>(qbf, kbf, vbf, obuf);
  k_proj_out<<<256, 256, 0, stream>>>(x, obuf, wbf + 3 * 4096, bp, (float*)d_out);
}
